// Sequence_26912265076994
// MI455X (gfx1250) — matrix operand order repaired and checked by reading, not compiled, not run on hardware
//
#include <hip/hip_runtime.h>
#include <hip/hip_bf16.h>

// ---------------------------------------------------------------------------
// MI455X (gfx1250) 2-layer LSTM LM, bf16 WMMA + async global->LDS staging.
//
// V=128 H=1024 E=256 C=128 R=512 B=256 T=512 FUT=64.
//
// Precompute folds (exact fp32, then one bf16 cast):
//   P[V,R]    = embed @ ee_w^T
//   Q[V,4H]   = P @ w_ih1^T            -> token path = per-row gather
//   ce2[4H,C] = w_ih1 @ ce_w           -> feature path folds into gate GEMM
//   bias1[4H] = b_ih1 + b_hh1 + (ee_b+ce_b) @ w_ih1^T
// Per step (weights resident in 192MB L2 -> compute/issue bound):
//   gates1 = bias1 + Q[tok] + feat_t @ ce2^T + h1 @ w_hh1^T   (K = 128+1024)
//   gates2 = bias2 + h1 @ w_ih2^T + h2 @ w_hh2^T              (K = 2048)
//   out_t  = h2 @ lin_w^T + lin_b                             (K = 1024)
//
// Each wave owns a 32(batch) x 16(j) x 4(gate) tile: 8 f32 accumulators,
// each weight B-fragment feeds 2 WMMAs. The shared 32-row A tile is staged
// once per block into LDS with GLOBAL_LOAD_ASYNC_TO_LDS_B128 (ASYNCcnt),
// then read as ds_load fragments; LSTM nonlinearity stays in-register.
// ---------------------------------------------------------------------------

typedef __attribute__((ext_vector_type(16))) __bf16 v16bf;
typedef __attribute__((ext_vector_type(8)))  __bf16 v8bf;
typedef __attribute__((ext_vector_type(8)))  float  v8f;
typedef int v4i_vec __attribute__((vector_size(16)));   // builtin's pointee type

#define WMMA_BF16(a, b, c) \
  __builtin_amdgcn_wmma_f32_16x16x32_bf16(false, (a), false, (b), (short)0, (c), false, false)

#if defined(__has_builtin)
#if __has_builtin(__builtin_amdgcn_global_load_async_to_lds_b128)
#define ASYNC_LDS 1
#else
#define ASYNC_LDS 0
#endif
#else
#define ASYNC_LDS 0
#endif

// One 16B chunk global -> LDS. Async path uses the CDNA5 ASYNC queue; the
// fallback is a synchronous register bounce (still kills the 8x per-wave
// redundancy on the A tile).
__device__ __forceinline__ void stage16(unsigned short* ldst, const unsigned short* gsrc) {
#if ASYNC_LDS
  __builtin_amdgcn_global_load_async_to_lds_b128(
      (__attribute__((address_space(1))) v4i_vec*)(const void*)gsrc,
      (__attribute__((address_space(3))) v4i_vec*)(void*)ldst,
      0, 0);
#else
  *reinterpret_cast<uint4*>(ldst) = *reinterpret_cast<const uint4*>(gsrc);
#endif
}

__device__ __forceinline__ void stage_wait() {
#if ASYNC_LDS
#if __has_builtin(__builtin_amdgcn_s_wait_asynccnt)
  __builtin_amdgcn_s_wait_asynccnt(0);
#else
  asm volatile("s_wait_asynccnt 0x0" ::: "memory");
#endif
#endif
  __syncthreads();
}

__device__ __forceinline__ unsigned short f32_to_bf16(float f) {
  unsigned int u = __float_as_uint(f);
  unsigned int r = u + 0x7FFFu + ((u >> 16) & 1u);   // round-to-nearest-even
  return (unsigned short)(r >> 16);
}

__device__ __forceinline__ float fast_sigmoid(float x) {
  return 1.f / (1.f + __expf(-x));                   // v_exp_f32 (TRANS)
}
__device__ __forceinline__ float fast_tanh(float x) {
  float e = __expf(-2.f * fabsf(x));
  float r = (1.f - e) / (1.f + e);
  return copysignf(r, x);
}

// A-fragment (16x32 bf16, ISA layout): lane<16 holds K {k..k+7, k+16..k+23},
// lane>=16 holds K {k+8..k+15, k+24..k+31} of row M = lane%16.
__device__ __forceinline__ v16bf load_a_frag(const unsigned short* rowptr, int k, int lhi) {
  union { v16bf v; v8bf h[2]; } u;
  const int o = k + lhi * 8;
  u.h[0] = *reinterpret_cast<const v8bf*>(rowptr + o);
  u.h[1] = *reinterpret_cast<const v8bf*>(rowptr + o + 16);
  return u.v;
}

// B-fragment (32x16 bf16): lane<16 holds K {k..k+7, k+16..k+23},
// lane>=16 holds K {k+8..k+15, k+24..k+31} of row N = lane%16.  W stored [N,K] row-major.
__device__ __forceinline__ v16bf load_b_frag(const unsigned short* rowptr, int k, int lhi) {
  union { v16bf v; v8bf h[2]; } u;
  const int o = k + lhi * 8;
  u.h[0] = *reinterpret_cast<const v8bf*>(rowptr + o);
  u.h[1] = *reinterpret_cast<const v8bf*>(rowptr + o + 16);
  return u.v;
}

// ------------------------------ precompute ---------------------------------

__global__ void cvt_bf16(const float* __restrict__ s, unsigned short* __restrict__ d, long long n) {
  long long i = (long long)blockIdx.x * blockDim.x + threadIdx.x;
  long long st = (long long)gridDim.x * blockDim.x;
  for (; i < n; i += st) d[i] = f32_to_bf16(s[i]);
}

__global__ void zero_u32(unsigned int* __restrict__ p, long long n) {
  long long i = (long long)blockIdx.x * blockDim.x + threadIdx.x;
  long long st = (long long)gridDim.x * blockDim.x;
  for (; i < n; i += st) p[i] = 0u;
}

// P[v*512+r] = sum_e embed[v,e] * ee_w[r,e]
__global__ void k_P(const float* __restrict__ embed, const float* __restrict__ eew,
                    float* __restrict__ P) {
  int id = blockIdx.x * 256 + threadIdx.x;
  int v = id >> 9, r = id & 511;
  const float* a = embed + (size_t)v * 256;
  const float* b = eew + (size_t)r * 256;
  float s = 0.f;
  for (int e = 0; e < 256; ++e) s += a[e] * b[e];
  P[id] = s;
}

// Q[v*4096+g] = sum_r P[v,r] * w_ih1[g,r]
__global__ void k_Q(const float* __restrict__ P, const float* __restrict__ wih1,
                    float* __restrict__ Q) {
  int id = blockIdx.x * 256 + threadIdx.x;
  int v = id >> 12, g = id & 4095;
  const float* a = P + (size_t)v * 512;
  const float* b = wih1 + (size_t)g * 512;
  float s = 0.f;
  for (int r = 0; r < 512; ++r) s += a[r] * b[r];
  Q[id] = s;
}

// ce2[g*128+c] = bf16( sum_r w_ih1[g,r] * ce_w[r,c] )
__global__ void k_ce2(const float* __restrict__ wih1, const float* __restrict__ cew,
                      unsigned short* __restrict__ ce2) {
  int id = blockIdx.x * 256 + threadIdx.x;
  int g = id >> 7, c = id & 127;
  const float* a = wih1 + (size_t)g * 512;
  float s = 0.f;
  for (int r = 0; r < 512; ++r) s += a[r] * cew[r * 128 + c];
  ce2[id] = f32_to_bf16(s);
}

__global__ void k_bias(const float* __restrict__ bih1, const float* __restrict__ bhh1,
                       const float* __restrict__ eeb,  const float* __restrict__ ceb,
                       const float* __restrict__ wih1,
                       const float* __restrict__ bih2, const float* __restrict__ bhh2,
                       float* __restrict__ bias1, float* __restrict__ bias2) {
  int g = blockIdx.x * 256 + threadIdx.x;
  if (g >= 4096) return;
  const float* a = wih1 + (size_t)g * 512;
  float s = bih1[g] + bhh1[g];
  for (int r = 0; r < 512; ++r) s += (eeb[r] + ceb[r]) * a[r];
  bias1[g] = s;
  bias2[g] = bih2[g] + bhh2[g];
}

// ------------------------------ per-step kernels ---------------------------
// 512 wave-tiles = 8 batch-tiles(32 rows) x 64 j-tiles, 8 waves/block
// -> 64 blocks; all 8 waves of a block share one 32-row A tile in LDS.

__global__ void __launch_bounds__(256) lstm1_step(
    const unsigned short* __restrict__ h1_old, unsigned short* __restrict__ h1_new,
    float* __restrict__ c1,
    const unsigned short* __restrict__ featbf,   // [256, 576, 128] bf16
    const unsigned short* __restrict__ ce2,      // [4096, 128]  bf16
    const unsigned short* __restrict__ whh1,     // [4096, 1024] bf16
    const float* __restrict__ Q,                 // [128, 4096]  f32
    const float* __restrict__ bias1,             // [4096]       f32
    const int* __restrict__ tok_base, int tok_mul, int tok_off, int t) {
  __shared__ unsigned short sh[32 * 1024];       // 64KB, reused per K-pass

  const int tid = threadIdx.x;
  const int lane = tid & 31, w = tid >> 5;
  const int wt = blockIdx.x * 8 + w;             // 0..511
  const int m0 = (wt >> 6) << 5;                 // batch-tile origin (32 rows)
  const int j0 = (wt & 63) << 4;                 // hidden-unit tile origin
  const int lmod = lane & 15, lhi = lane >> 4;

  v8f acc[8];
#pragma unroll
  for (int g = 0; g < 8; ++g)
    acc[g] = v8f{0.f, 0.f, 0.f, 0.f, 0.f, 0.f, 0.f, 0.f};

  // -------- pass 1: conditioning features (K=128, B = w_ih1 @ ce_w) --------
  for (int c = tid; c < 512; c += 256) {         // 32 rows x 256B
    int row = c >> 4, col = (c & 15) << 3;
    stage16(sh + row * 128 + col,
            featbf + ((size_t)(m0 + row) * 576 + t) * 128 + col);
  }
  stage_wait();
  {
    const unsigned short* A0 = sh + (size_t)lmod * 128;
    const unsigned short* A1 = sh + (size_t)(16 + lmod) * 128;
    const unsigned short* Brow[4];
#pragma unroll
    for (int g = 0; g < 4; ++g) Brow[g] = ce2 + (size_t)(g * 1024 + j0 + lmod) * 128;
    for (int k = 0; k < 128; k += 32) {
      v16bf a0 = load_a_frag(A0, k, lhi);
      v16bf a1 = load_a_frag(A1, k, lhi);
#pragma unroll
      for (int g = 0; g < 4; ++g) {
        v16bf b = load_b_frag(Brow[g], k, lhi);
        acc[g]     = WMMA_BF16(a0, b, acc[g]);
        acc[4 + g] = WMMA_BF16(a1, b, acc[4 + g]);
      }
    }
  }
  __syncthreads();                               // all reads done before restage

  // -------- pass 2: recurrent h1 @ w_hh1^T (K=1024) --------
  {
    const unsigned short* hsrc = h1_old + (size_t)m0 * 1024;  // contiguous 64KB
    for (int c = tid; c < 4096; c += 256) stage16(sh + c * 8, hsrc + c * 8);
    stage_wait();
    const unsigned short* A0 = sh + (size_t)lmod * 1024;
    const unsigned short* A1 = sh + (size_t)(16 + lmod) * 1024;
    const unsigned short* Brow[4];
#pragma unroll
    for (int g = 0; g < 4; ++g) Brow[g] = whh1 + (size_t)(g * 1024 + j0 + lmod) * 1024;
    for (int k = 0; k < 1024; k += 32) {
      v16bf a0 = load_a_frag(A0, k, lhi);
      v16bf a1 = load_a_frag(A1, k, lhi);
#pragma unroll
      for (int g = 0; g < 4; ++g) {
        v16bf b = load_b_frag(Brow[g], k, lhi);
        acc[g]     = WMMA_BF16(a0, b, acc[g]);
        acc[4 + g] = WMMA_BF16(a1, b, acc[4 + g]);
      }
    }
  }

  // -------- in-register LSTM nonlinearity --------
  // D layout: row = m0 + part*16 + lhi*8 + r, col j = j0 + lane%16.
  const int j = j0 + lmod;
#pragma unroll
  for (int part = 0; part < 2; ++part) {
#pragma unroll
    for (int r = 0; r < 8; ++r) {
      const int b = m0 + part * 16 + lhi * 8 + r;
      const int tok = tok_base[b * tok_mul + tok_off];
      const float* q = Q + (size_t)tok * 4096;
      float gi = acc[part * 4 + 0][r] + bias1[j]        + q[j];
      float gf = acc[part * 4 + 1][r] + bias1[1024 + j] + q[1024 + j];
      float gg = acc[part * 4 + 2][r] + bias1[2048 + j] + q[2048 + j];
      float go = acc[part * 4 + 3][r] + bias1[3072 + j] + q[3072 + j];
      const size_t idx = (size_t)b * 1024 + j;
      float cn = fast_sigmoid(gf) * c1[idx] + fast_sigmoid(gi) * fast_tanh(gg);
      c1[idx] = cn;
      h1_new[idx] = f32_to_bf16(fast_sigmoid(go) * fast_tanh(cn));
    }
  }
}

__global__ void __launch_bounds__(256) lstm2_step(
    const unsigned short* __restrict__ h1_new,
    const unsigned short* __restrict__ h2_old, unsigned short* __restrict__ h2_new,
    float* __restrict__ c2,
    const unsigned short* __restrict__ wih2, const unsigned short* __restrict__ whh2,
    const float* __restrict__ bias2) {
  __shared__ unsigned short sh[32 * 1024];       // 64KB, reused per K-pass

  const int tid = threadIdx.x;
  const int lane = tid & 31, w = tid >> 5;
  const int wt = blockIdx.x * 8 + w;
  const int m0 = (wt >> 6) << 5;
  const int j0 = (wt & 63) << 4;
  const int lmod = lane & 15, lhi = lane >> 4;

  v8f acc[8];
#pragma unroll
  for (int g = 0; g < 8; ++g)
    acc[g] = v8f{0.f, 0.f, 0.f, 0.f, 0.f, 0.f, 0.f, 0.f};

  const unsigned short* Apass[2] = { h1_new + (size_t)m0 * 1024,
                                     h2_old + (size_t)m0 * 1024 };
  const unsigned short* Wpass[2] = { wih2, whh2 };

  for (int pass = 0; pass < 2; ++pass) {
    if (pass) __syncthreads();                   // reads of prev tile done
    for (int c = tid; c < 4096; c += 256) stage16(sh + c * 8, Apass[pass] + c * 8);
    stage_wait();
    const unsigned short* A0 = sh + (size_t)lmod * 1024;
    const unsigned short* A1 = sh + (size_t)(16 + lmod) * 1024;
    const unsigned short* Brow[4];
#pragma unroll
    for (int g = 0; g < 4; ++g)
      Brow[g] = Wpass[pass] + (size_t)(g * 1024 + j0 + lmod) * 1024;
    for (int k = 0; k < 1024; k += 32) {
      v16bf a0 = load_a_frag(A0, k, lhi);
      v16bf a1 = load_a_frag(A1, k, lhi);
#pragma unroll
      for (int g = 0; g < 4; ++g) {
        v16bf b = load_b_frag(Brow[g], k, lhi);
        acc[g]     = WMMA_BF16(a0, b, acc[g]);
        acc[4 + g] = WMMA_BF16(a1, b, acc[4 + g]);
      }
    }
  }

  const int j = j0 + lmod;
#pragma unroll
  for (int part = 0; part < 2; ++part) {
#pragma unroll
    for (int r = 0; r < 8; ++r) {
      const int b = m0 + part * 16 + lhi * 8 + r;
      float gi = acc[part * 4 + 0][r] + bias2[j];
      float gf = acc[part * 4 + 1][r] + bias2[1024 + j];
      float gg = acc[part * 4 + 2][r] + bias2[2048 + j];
      float go = acc[part * 4 + 3][r] + bias2[3072 + j];
      const size_t idx = (size_t)b * 1024 + j;
      float cn = fast_sigmoid(gf) * c2[idx] + fast_sigmoid(gi) * fast_tanh(gg);
      c2[idx] = cn;
      h2_new[idx] = f32_to_bf16(fast_sigmoid(go) * fast_tanh(cn));
    }
  }
}

// Logits: out[:, t, :] = h2 @ lin_w^T + lin_b.
// 128 wave-tiles (16 batch x 8 vocab) = 16 blocks; block shares one 16-row
// 32KB A tile in LDS.
__global__ void __launch_bounds__(256) out_step(
    const unsigned short* __restrict__ h2_new,
    const unsigned short* __restrict__ linbf,   // [128, 1024] bf16
    const float* __restrict__ lin_b,
    float* __restrict__ out, int t) {
  __shared__ unsigned short sh[16 * 1024];      // 32KB

  const int tid = threadIdx.x;
  const int lane = tid & 31, w = tid >> 5;
  const int m0 = blockIdx.x << 4;               // wt>>3 == blockIdx
  const int n0 = w << 4;
  const int lmod = lane & 15, lhi = lane >> 4;

  const unsigned short* hsrc = h2_new + (size_t)m0 * 1024;  // contiguous 32KB
  for (int c = tid; c < 2048; c += 256) stage16(sh + c * 8, hsrc + c * 8);
  stage_wait();

  const unsigned short* Arow = sh + (size_t)lmod * 1024;
  const unsigned short* Brow = linbf + (size_t)(n0 + lmod) * 1024;
  v8f acc = v8f{0.f, 0.f, 0.f, 0.f, 0.f, 0.f, 0.f, 0.f};
  for (int k = 0; k < 1024; k += 32) {
    v16bf a = load_a_frag(Arow, k, lhi);
    v16bf b = load_b_frag(Brow, k, lhi);
    acc = WMMA_BF16(a, b, acc);
  }
  const int n = n0 + lmod;
  const float bias = lin_b[n];
#pragma unroll
  for (int r = 0; r < 8; ++r) {
    const int b = m0 + lhi * 8 + r;
    out[((size_t)b * 576 + t) * 128 + n] = acc[r] + bias;
  }
}

// Autoregressive token selection (first-max semantics like jnp.argmax).
__global__ void argmax_step(const float* __restrict__ out, int tprev,
                            int* __restrict__ tokcur) {
  int b = threadIdx.x;  // <<<1,256>>>
  const float* row = out + ((size_t)b * 576 + tprev) * 128;
  float best = row[0];
  int bi = 0;
  for (int v = 1; v < 128; ++v) {
    float x = row[v];
    if (x > best) { best = x; bi = v; }
  }
  tokcur[b] = bi;
}

// ------------------------------ host driver --------------------------------

extern "C" void kernel_launch(void* const* d_in, const int* in_sizes, int n_in,
                              void* d_out, int out_size, void* d_ws, size_t ws_size,
                              hipStream_t stream) {
  (void)in_sizes; (void)n_in; (void)out_size; (void)ws_size;
  const int*   tokens   = (const int*)  d_in[0];
  const float* features = (const float*)d_in[1];
  // d_in[2] = future (compile-time 64)
  const float* embed_W = (const float*)d_in[3];
  const float* ee_w    = (const float*)d_in[4];
  const float* ee_b    = (const float*)d_in[5];
  const float* ce_w    = (const float*)d_in[6];
  const float* ce_b    = (const float*)d_in[7];
  const float* w_ih1   = (const float*)d_in[8];
  const float* w_hh1   = (const float*)d_in[9];
  const float* b_ih1   = (const float*)d_in[10];
  const float* b_hh1   = (const float*)d_in[11];
  const float* w_ih2   = (const float*)d_in[12];
  const float* w_hh2   = (const float*)d_in[13];
  const float* b_ih2   = (const float*)d_in[14];
  const float* b_hh2   = (const float*)d_in[15];
  const float* lin_w   = (const float*)d_in[16];
  const float* lin_b   = (const float*)d_in[17];
  float* out = (float*)d_out;

  constexpr int T = 512, FUT = 64, TF = 576;

  // Bump allocator over d_ws (256B aligned regions).  ~71 MB total.
  char* ws = (char*)d_ws;
  size_t off = 0;
  auto alloc = [&](size_t bytes) -> char* {
    char* p = ws + off;
    off += bytes;
    off = (off + 255) & ~(size_t)255;
    return p;
  };
  unsigned short* whh1_bf = (unsigned short*)alloc((size_t)4096 * 1024 * 2);
  unsigned short* wih2_bf = (unsigned short*)alloc((size_t)4096 * 1024 * 2);
  unsigned short* whh2_bf = (unsigned short*)alloc((size_t)4096 * 1024 * 2);
  unsigned short* lin_bf  = (unsigned short*)alloc((size_t)128 * 1024 * 2);
  unsigned short* ce2_bf  = (unsigned short*)alloc((size_t)4096 * 128 * 2);
  unsigned short* feat_bf = (unsigned short*)alloc((size_t)256 * TF * 128 * 2);
  float* P     = (float*)alloc((size_t)128 * 512 * 4);
  float* Q     = (float*)alloc((size_t)128 * 4096 * 4);
  float* bias1 = (float*)alloc(4096 * 4);
  float* bias2 = (float*)alloc(4096 * 4);
  size_t state_begin = off;
  unsigned short* h1A = (unsigned short*)alloc((size_t)256 * 1024 * 2);
  unsigned short* h1B = (unsigned short*)alloc((size_t)256 * 1024 * 2);
  unsigned short* h2A = (unsigned short*)alloc((size_t)256 * 1024 * 2);
  unsigned short* h2B = (unsigned short*)alloc((size_t)256 * 1024 * 2);
  float* c1 = (float*)alloc((size_t)256 * 1024 * 4);
  float* c2 = (float*)alloc((size_t)256 * 1024 * 4);
  int* tokcur = (int*)alloc(256 * 4);
  size_t state_end = off;

  // --- one-time precompute (re-run each call; idempotent, reads only d_in) ---
  cvt_bf16<<<2048, 256, 0, stream>>>(w_hh1, whh1_bf, (long long)4096 * 1024);
  cvt_bf16<<<2048, 256, 0, stream>>>(w_ih2, wih2_bf, (long long)4096 * 1024);
  cvt_bf16<<<2048, 256, 0, stream>>>(w_hh2, whh2_bf, (long long)4096 * 1024);
  cvt_bf16<<<256,  256, 0, stream>>>(lin_w, lin_bf,  (long long)128 * 1024);
  cvt_bf16<<<4096, 256, 0, stream>>>(features, feat_bf, (long long)256 * TF * 128);
  k_P  <<<256,  256, 0, stream>>>(embed_W, ee_w, P);
  k_Q  <<<2048, 256, 0, stream>>>(P, w_ih1, Q);
  k_ce2<<<2048, 256, 0, stream>>>(w_ih1, ce_w, ce2_bf);
  k_bias<<<16,  256, 0, stream>>>(b_ih1, b_hh1, ee_b, ce_b, w_ih1, b_ih2, b_hh2,
                                  bias1, bias2);
  zero_u32<<<1024, 256, 0, stream>>>((unsigned int*)(ws + state_begin),
                                     (long long)((state_end - state_begin) / 4));

  // --- sequential recurrence: 512 teacher-forced + 64 autoregressive steps ---
  for (int t = 0; t < T + FUT; ++t) {
    const unsigned short* h1o = (t & 1) ? h1B : h1A;
    unsigned short*       h1n = (t & 1) ? h1A : h1B;
    const unsigned short* h2o = (t & 1) ? h2B : h2A;
    unsigned short*       h2n = (t & 1) ? h2A : h2B;

    const int* tok_base; int tok_mul, tok_off;
    if (t < T) {
      tok_base = tokens; tok_mul = T; tok_off = t;
    } else {
      argmax_step<<<1, 256, 0, stream>>>(out, t - 1, tokcur);
      tok_base = tokcur; tok_mul = 1; tok_off = 0;
    }

    lstm1_step<<<64, 256, 0, stream>>>(h1o, h1n, c1, feat_bf, ce2_bf, whh1_bf,
                                       Q, bias1, tok_base, tok_mul, tok_off, t);
    lstm2_step<<<64, 256, 0, stream>>>(h1n, h2o, h2n, c2, wih2_bf, whh2_bf, bias2);
    out_step  <<<16, 256, 0, stream>>>(h2n, lin_bf, lin_b, out, t);
  }
}
